// DoubleAttention_82703890252117
// MI455X (gfx1250) — matrix operand order repaired and checked by reading, not compiled, not run on hardware
//
#include <hip/hip_runtime.h>
#include <hip/hip_bf16.h>
#include <stdint.h>

typedef __bf16 bf16_t;
typedef __attribute__((ext_vector_type(16))) __bf16 v16bf;
typedef __attribute__((ext_vector_type(8)))  float  v8f;
typedef __attribute__((ext_vector_type(4)))  unsigned int u32x4;
typedef __attribute__((ext_vector_type(2)))  unsigned int u32x2;
typedef __attribute__((ext_vector_type(4)))  float  f32x4;

union FragBF { v16bf v; u32x4 q[2]; };
union Pack4  { bf16_t h[4]; u32x2 d; };
union Pack8  { bf16_t h[8]; u32x4 d; };

// CDNA5 async global->LDS copy (ASYNCcnt-tracked, no VGPR round trip).
// Generic (flat) pointers to __shared__ carry the LDS offset in addr[31:0]
// (ISA 10.2 aperture rule), so the low 32 bits are the VDST LDS address.
__device__ __forceinline__ void async_copy_b128(unsigned lds_off, const void* gaddr) {
    asm volatile("global_load_async_to_lds_b128 %0, %1, off"
                 :: "v"(lds_off), "v"(gaddr) : "memory");
}
__device__ __forceinline__ void wait_async0() {
    asm volatile("s_wait_asynccnt 0x0" ::: "memory");
}

enum { OUT_F32_BIAS = 0, OUT_BF16_BIAS = 1, OUT_BF16 = 2, OUT_BF16_T_BIAS = 3, OUT_F32_T = 4 };

// Tiled WMMA GEMM, async double-buffered staging.
//   D[M,N] = A[M,K] * B(K,N),  A: bf16 row-major (ldA), B: bf16 [N][K] (ldB)
//   grid.z = batch*zdiv + kchunk (split-K); C offset = z*sC
// Block: 128x128 tile, BK=64, 256 threads (8 waves), wave tile 64x32 -> 16 WMMA/step
template <int OUT_MODE>
__global__ __launch_bounds__(256) void gemm_async_wmma(
    const bf16_t* __restrict__ Ap, const bf16_t* __restrict__ Bp,
    const float* __restrict__ bias, void* __restrict__ Cp,
    int M, int N, int K, int ldA, int ldB,
    long long sA, long long sB, long long sC,
    int zdiv, long long aKoff, long long bKoff)
{
    __shared__ __align__(16) bf16_t As[2][128][72];   // pad 64->72 (144B rows, 16B aligned)
    __shared__ __align__(16) bf16_t Bs[2][128][72];   // stored as B^T : [n][k]

    const int tid  = threadIdx.x;
    const int lane = tid & 31;
    const int wid  = tid >> 5;
    const int wave_m = wid & 1;
    const int wave_n = wid >> 1;
    const int m0 = blockIdx.y * 128;
    const int n0 = blockIdx.x * 128;
    const int z  = blockIdx.z;
    const int batch = z / zdiv;
    const int kc    = z - batch * zdiv;

    const bf16_t* Ab = Ap + (long long)batch * sA + (long long)kc * aKoff;
    const bf16_t* Bb = Bp + (long long)batch * sB + (long long)kc * bKoff;

    auto stage = [&](int buf, int ks) {
#pragma unroll
        for (int i = 0; i < 4; i++) {           // A tile: 128 x 64 bf16 = 1024 x 16B
            int idx = tid + i * 256;
            int row = idx >> 3, c8 = idx & 7;
            async_copy_b128((unsigned)(uintptr_t)&As[buf][row][c8 * 8],
                            Ab + (long long)(m0 + row) * ldA + ks + c8 * 8);
        }
#pragma unroll
        for (int i = 0; i < 4; i++) {           // B tile: 128 x 64 bf16
            int idx = tid + i * 256;
            int row = idx >> 3, c8 = idx & 7;
            async_copy_b128((unsigned)(uintptr_t)&Bs[buf][row][c8 * 8],
                            Bb + (long long)(n0 + row) * ldB + ks + c8 * 8);
        }
    };

    v8f acc[4][2];
    const v8f vzero = {0.f, 0.f, 0.f, 0.f, 0.f, 0.f, 0.f, 0.f};
#pragma unroll
    for (int i = 0; i < 4; i++)
#pragma unroll
        for (int j = 0; j < 2; j++) acc[i][j] = vzero;

    const int mrow = wave_m * 64 + (lane & 15);
    const int kba  = (lane < 16) ? 0 : 8;      // A lane k-chunks {kba..+7, kba+16..+23}
    const int nrow = wave_n * 32 + (lane & 15);
    const int kbb  = (lane < 16) ? 0 : 8;      // B lane k-chunks {kbb..+7, kbb+16..+23}

    const int S = K >> 6;
    stage(0, 0);
    wait_async0();
    __syncthreads();

    for (int s = 0; s < S; s++) {
        const int buf = s & 1;
        if (s + 1 < S) stage(buf ^ 1, (s + 1) << 6);   // async fills other buffer during compute
#pragma unroll
        for (int kk = 0; kk < 64; kk += 32) {
            FragBF a[4], b[2];
#pragma unroll
            for (int i = 0; i < 4; i++) {
                a[i].q[0] = *(const u32x4*)&As[buf][mrow + i * 16][kk + kba];
                a[i].q[1] = *(const u32x4*)&As[buf][mrow + i * 16][kk + kba + 16];
            }
#pragma unroll
            for (int j = 0; j < 2; j++) {
                b[j].q[0] = *(const u32x4*)&Bs[buf][nrow + j * 16][kk + kbb];
                b[j].q[1] = *(const u32x4*)&Bs[buf][nrow + j * 16][kk + kbb + 16];
            }
#pragma unroll
            for (int i = 0; i < 4; i++)
#pragma unroll
                for (int j = 0; j < 2; j++)
                    acc[i][j] = __builtin_amdgcn_wmma_f32_16x16x32_bf16(
                        false, a[i].v, false, b[j].v, (short)0, acc[i][j], false, false);
        }
        if (s + 1 < S) { wait_async0(); __syncthreads(); }
    }

    // epilogue: VGPR g -> row g (+8 for lanes 16..31), col = lane&15
#pragma unroll
    for (int i = 0; i < 4; i++) {
        const int rbase = m0 + wave_m * 64 + i * 16 + ((lane < 16) ? 0 : 8);
#pragma unroll
        for (int j = 0; j < 2; j++) {
            const int cg = n0 + wave_n * 32 + j * 16 + (lane & 15);
            if constexpr (OUT_MODE == OUT_F32_BIAS) {
                float* C = (float*)Cp + (long long)z * sC;
                const float bb = bias[cg];
#pragma unroll
                for (int g = 0; g < 8; g++)
                    C[(long long)(rbase + g) * N + cg] = acc[i][j][g] + bb;
            } else if constexpr (OUT_MODE == OUT_BF16_BIAS) {
                bf16_t* C = (bf16_t*)Cp + (long long)z * sC;
                const float bb = bias[cg];
#pragma unroll
                for (int g = 0; g < 8; g++)
                    C[(long long)(rbase + g) * N + cg] = (bf16_t)(acc[i][j][g] + bb);
            } else if constexpr (OUT_MODE == OUT_BF16) {
                bf16_t* C = (bf16_t*)Cp + (long long)z * sC;
#pragma unroll
                for (int g = 0; g < 8; g++)
                    C[(long long)(rbase + g) * N + cg] = (bf16_t)acc[i][j][g];
            } else if constexpr (OUT_MODE == OUT_BF16_T_BIAS) {
                bf16_t* C = (bf16_t*)Cp + (long long)z * sC;    // C[col][row], rows contiguous
                const float bb = bias[cg];
                Pack8 pk;
#pragma unroll
                for (int g = 0; g < 8; g++) pk.h[g] = (bf16_t)(acc[i][j][g] + bb);
                *(u32x4*)&C[(long long)cg * M + rbase] = pk.d;
            } else {  // OUT_F32_T: f32 transposed partial (split-K)
                float* C = (float*)Cp + (long long)z * sC;
                f32x4 lo = {acc[i][j][0], acc[i][j][1], acc[i][j][2], acc[i][j][3]};
                f32x4 hi = {acc[i][j][4], acc[i][j][5], acc[i][j][6], acc[i][j][7]};
                *(f32x4*)&C[(long long)cg * M + rbase]     = lo;
                *(f32x4*)&C[(long long)cg * M + rbase + 4] = hi;
            }
        }
    }
}

// ---- x : f32 -> bf16, vectorized x4 ----
__global__ __launch_bounds__(256) void convert_f32_bf16_kernel(
    const float* __restrict__ src, bf16_t* __restrict__ dst, long long n4)
{
    long long t = (long long)blockIdx.x * 256 + threadIdx.x;
    if (t >= n4) return;
    f32x4 v = *(const f32x4*)&src[t * 4];
    Pack4 pk;
    pk.h[0] = (bf16_t)v.x; pk.h[1] = (bf16_t)v.y;
    pk.h[2] = (bf16_t)v.z; pk.h[3] = (bf16_t)v.w;
    *(u32x2*)&dst[t * 4] = pk.d;
}

// ---- weights: dst[j][i] = (bf16)src[i][j] ----
__global__ __launch_bounds__(256) void transpose_w_kernel(
    const float* __restrict__ src, bf16_t* __restrict__ dst, int R, int Ccols)
{
    long long t = (long long)blockIdx.x * 256 + threadIdx.x;
    if (t >= (long long)R * Ccols) return;
    int i = (int)(t / Ccols), j = (int)(t % Ccols);
    dst[(long long)j * R + i] = (bf16_t)src[t];
}

// ---- softmax over n, phase 1: per-chunk column max & sum(exp(.-max)) ----
// grid (V/64, B, nch); block 256 = 64 channels x 4 row-parallel
__global__ __launch_bounds__(256) void softmax_partial_kernel(
    const bf16_t* __restrict__ src, float* __restrict__ Pmax, float* __restrict__ Psum,
    int Nrows, int V, int nch)
{
    __shared__ float redm[4][64], reds[4][64];
    const int tid = threadIdx.x, vt = tid & 63, nt = tid >> 6;
    const int v0 = blockIdx.x * 64, v = v0 + vt;
    const int b = blockIdx.y, ch = blockIdx.z;
    const int rows = Nrows / nch, nbeg = ch * rows;
    const bf16_t* s = src + (long long)b * Nrows * V;

    float m = -3.0e38f;
    for (int n = nbeg + nt; n < nbeg + rows; n += 4)
        m = fmaxf(m, (float)s[(long long)n * V + v]);
    redm[nt][vt] = m;
    __syncthreads();
    const float cm = fmaxf(fmaxf(redm[0][vt], redm[1][vt]),
                           fmaxf(redm[2][vt], redm[3][vt]));
    float ss = 0.f;
    for (int n = nbeg + nt; n < nbeg + rows; n += 4)
        ss += __expf((float)s[(long long)n * V + v] - cm);
    reds[nt][vt] = ss;
    __syncthreads();
    if (tid < 64) {
        long long o = ((long long)b * nch + ch) * V + v0 + tid;
        Pmax[o] = cm;
        Psum[o] = reds[0][tid] + reds[1][tid] + reds[2][tid] + reds[3][tid];
    }
}

// ---- softmax phase 2: merge chunk partials (online max/sum) ----
__global__ __launch_bounds__(256) void softmax_combine_kernel(
    const float* __restrict__ Pmax, const float* __restrict__ Psum,
    float* __restrict__ Cmax, float* __restrict__ Cinv, int V, int nch, int Bt)
{
    int t = blockIdx.x * 256 + threadIdx.x;
    if (t >= Bt * V) return;
    int b = t / V, v = t - b * V;
    float m = -3.0e38f;
    for (int ch = 0; ch < nch; ch++)
        m = fmaxf(m, Pmax[((long long)b * nch + ch) * V + v]);
    float s = 0.f;
    for (int ch = 0; ch < nch; ch++) {
        long long o = ((long long)b * nch + ch) * V + v;
        s += Psum[o] * __expf(Pmax[o] - m);
    }
    Cmax[t] = m;
    Cinv[t] = 1.0f / s;
}

// ---- softmax phase 3: normalize; TRANS writes [b][v][n] via LDS transpose ----
template <bool TRANS>
__global__ __launch_bounds__(256) void softmax_apply_kernel(
    const bf16_t* __restrict__ src, bf16_t* __restrict__ dst,
    const float* __restrict__ Cmax, const float* __restrict__ Cinv,
    int Nrows, int V, int nch)
{
    __shared__ __align__(16) bf16_t tile[64][80];
    const int tid = threadIdx.x, vt = tid & 63, nt = tid >> 6;
    const int v0 = blockIdx.x * 64, v = v0 + vt;
    const int b = blockIdx.y, ch = blockIdx.z;
    const int rows = Nrows / nch, nbeg = ch * rows;
    const bf16_t* s = src + (long long)b * Nrows * V;
    dst += (long long)b * Nrows * V;
    const float mx = Cmax[(long long)b * V + v];
    const float ci = Cinv[(long long)b * V + v];

    if constexpr (!TRANS) {
        for (int n = nbeg + nt; n < nbeg + rows; n += 4)
            dst[(long long)n * V + v] =
                (bf16_t)(__expf((float)s[(long long)n * V + v] - mx) * ci);
    } else {
        for (int n0 = nbeg; n0 < nbeg + rows; n0 += 64) {
            for (int r = nt; r < 64; r += 4)
                tile[vt][r] = (bf16_t)(
                    __expf((float)s[(long long)(n0 + r) * V + v] - mx) * ci);
            __syncthreads();
            const int rr = tid >> 2, seg = tid & 3;   // 64 rows x 4 x 32B out
            u32x4* dp = (u32x4*)&dst[(long long)(v0 + rr) * Nrows + n0 + seg * 16];
            const u32x4* sp = (const u32x4*)&tile[rr][seg * 16];
            dp[0] = sp[0];
            dp[1] = sp[1];
            __syncthreads();
        }
    }
}

// ---- split-K reduction: Gt[b][e] = (bf16) sum_c Gp[b*nch+c][e] ----
__global__ __launch_bounds__(256) void reduce_g_kernel(
    const float* __restrict__ Gp, bf16_t* __restrict__ Gt,
    int perBatch, int nch, long long total)
{
    long long t = (long long)blockIdx.x * 256 + threadIdx.x;
    if (t >= total) return;
    long long b = t / perBatch, e = t - b * perBatch;
    float s = 0.f;
    for (int c = 0; c < nch; c++)
        s += Gp[((long long)b * nch + c) * perBatch + e];
    Gt[t] = (bf16_t)s;
}

extern "C" void kernel_launch(void* const* d_in, const int* in_sizes, int n_in,
                              void* d_out, int out_size, void* d_ws, size_t ws_size,
                              hipStream_t stream)
{
    (void)in_sizes; (void)n_in; (void)out_size; (void)ws_size;
    const float* x  = (const float*)d_in[0];
    const float* Wq = (const float*)d_in[1];
    const float* bq = (const float*)d_in[2];
    const float* Wk = (const float*)d_in[3];
    const float* bk = (const float*)d_in[4];
    const float* Wv = (const float*)d_in[5];
    const float* bv = (const float*)d_in[6];
    const float* Wr = (const float*)d_in[7];
    const float* br = (const float*)d_in[8];
    float* out = (float*)d_out;

    const int Bb = 16, Cc = 512, DIMd = 256, VD = 256;
    const int Np = 64 * 64;                         // 4096
    const long long Mtot = (long long)Bb * Np;      // 65536
    const int NCH = 8;                              // softmax n-chunks
    const int KSPLIT = 8, Kc = Np / KSPLIT;         // 512

    char* ws = (char*)d_ws;
    size_t off = 0;
    auto carve = [&](size_t bytes) {
        char* p = ws + off;
        off += (bytes + 255) & ~(size_t)255;
        return p;
    };
    bf16_t* wqT  = (bf16_t*)carve((size_t)Cc * DIMd * 2);   // [DIM][C]
    bf16_t* wkT  = (bf16_t*)carve((size_t)Cc * VD * 2);     // [VD][C]
    bf16_t* wvT  = (bf16_t*)carve((size_t)Cc * VD * 2);     // [VD][C]
    bf16_t* wrT  = (bf16_t*)carve((size_t)DIMd * Cc * 2);   // [C][DIM]
    bf16_t* qt   = (bf16_t*)carve((size_t)DIMd * Mtot * 2); // [d][b*n]
    bf16_t* kb   = (bf16_t*)carve((size_t)Mtot * VD * 2);   // [b*n][v]
    bf16_t* vb   = (bf16_t*)carve((size_t)Mtot * VD * 2);   // [b*n][v]
    bf16_t* At   = (bf16_t*)carve((size_t)Mtot * VD * 2);   // [b][v][n]
    bf16_t* Avm  = (bf16_t*)carve((size_t)Mtot * VD * 2);   // [b][n][v]
    float*  Pmax = (float*)carve((size_t)Bb * NCH * VD * 4);
    float*  Psum = (float*)carve((size_t)Bb * NCH * VD * 4);
    float*  Cmx  = (float*)carve((size_t)Bb * VD * 4);
    float*  Cin  = (float*)carve((size_t)Bb * VD * 4);
    bf16_t* xb = At;          // alias: 64MB (At+Avm), dead before softmax writes At/Avm
    float*  Gp = (float*)kb;  // alias: split-K partials; k dead after its softmax
    bf16_t* Gt = vb;          // alias: v dead after its softmax
    bf16_t* o2 = kb;          // alias: written after Gp consumed

    const dim3 blk(256);

    // 0) x -> bf16 ; weights -> transposed bf16
    {
        long long n4 = Mtot * Cc / 4;
        convert_f32_bf16_kernel<<<dim3((unsigned)((n4 + 255) / 256)), blk, 0, stream>>>(x, xb, n4);
        transpose_w_kernel<<<dim3((Cc * DIMd + 255) / 256), blk, 0, stream>>>(Wq, wqT, Cc, DIMd);
        transpose_w_kernel<<<dim3((Cc * VD + 255) / 256),   blk, 0, stream>>>(Wk, wkT, Cc, VD);
        transpose_w_kernel<<<dim3((Cc * VD + 255) / 256),   blk, 0, stream>>>(Wv, wvT, Cc, VD);
        transpose_w_kernel<<<dim3((DIMd * Cc + 255) / 256), blk, 0, stream>>>(Wr, wrT, DIMd, Cc);
    }

    // 1) projections: [65536 x 512] * [512 x 256] + bias
    {
        dim3 g(DIMd / 128, (unsigned)(Mtot / 128), 1);
        gemm_async_wmma<OUT_BF16_T_BIAS><<<g, blk, 0, stream>>>(   // q stored [d][b*n]
            xb, wqT, bq, qt, (int)Mtot, DIMd, Cc, Cc, Cc, 0, 0, 0, 1, 0, 0);
        gemm_async_wmma<OUT_BF16_BIAS><<<g, blk, 0, stream>>>(
            xb, wkT, bk, kb, (int)Mtot, VD, Cc, Cc, Cc, 0, 0, 0, 1, 0, 0);
        gemm_async_wmma<OUT_BF16_BIAS><<<g, blk, 0, stream>>>(
            xb, wvT, bv, vb, (int)Mtot, VD, Cc, Cc, Cc, 0, 0, 0, 1, 0, 0);
    }

    // 2) column softmax over positions: k -> At [b][v][n], v -> Avm [b][n][v]
    softmax_partial_kernel<<<dim3(VD / 64, Bb, NCH), blk, 0, stream>>>(kb, Pmax, Psum, Np, VD, NCH);
    softmax_combine_kernel<<<dim3((Bb * VD + 255) / 256), blk, 0, stream>>>(Pmax, Psum, Cmx, Cin, VD, NCH, Bb);
    softmax_apply_kernel<true><<<dim3(VD / 64, Bb, NCH), blk, 0, stream>>>(kb, At, Cmx, Cin, Np, VD, NCH);

    softmax_partial_kernel<<<dim3(VD / 64, Bb, NCH), blk, 0, stream>>>(vb, Pmax, Psum, Np, VD, NCH);
    softmax_combine_kernel<<<dim3((Bb * VD + 255) / 256), blk, 0, stream>>>(Pmax, Psum, Cmx, Cin, VD, NCH, Bb);
    softmax_apply_kernel<false><<<dim3(VD / 64, Bb, NCH), blk, 0, stream>>>(vb, Avm, Cmx, Cin, Np, VD, NCH);

    // 3) G[v,d] = sum_n At[v][n] q[n][d], split-K x8 -> f32 partials Gp[z][d][v]
    gemm_async_wmma<OUT_F32_T><<<dim3(DIMd / 128, VD / 128, Bb * KSPLIT), blk, 0, stream>>>(
        At, qt, nullptr, Gp, VD, DIMd, Kc, Np, (int)Mtot,
        (long long)VD * Np, (long long)Np, (long long)DIMd * VD,
        KSPLIT, (long long)Kc, (long long)Kc);
    reduce_g_kernel<<<dim3((unsigned)(((long long)Bb * DIMd * VD + 255) / 256)), blk, 0, stream>>>(
        Gp, Gt, DIMd * VD, KSPLIT, (long long)Bb * DIMd * VD);

    // 4) out2[n,d] = sum_v Avm[n][v] * Gt[d][v]   (per batch)
    gemm_async_wmma<OUT_BF16><<<dim3(DIMd / 128, Np / 128, Bb), blk, 0, stream>>>(
        Avm, Gt, nullptr, o2, Np, DIMd, VD, VD, VD,
        (long long)Np * VD, (long long)DIMd * VD, (long long)Np * DIMd, 1, 0, 0);

    // 5) y = out2 * Wr + br : [65536 x 256] * [256 x 512] -> f32 d_out
    gemm_async_wmma<OUT_F32_BIAS><<<dim3(Cc / 128, (unsigned)(Mtot / 128), 1), blk, 0, stream>>>(
        o2, wrT, br, out, (int)Mtot, Cc, DIMd, DIMd, DIMd, 0, 0, 0, 1, 0, 0);
}
